// GCN_22428319219871
// MI455X (gfx1250) — compile-verified, matrix operand order checked by reading, not run on hardware
//
#include <hip/hip_runtime.h>

// GCN forward (2x GCNConv + linear + softmax) for MI455X (gfx1250, wave32).
// x@W1 and h1@W2 run on the WMMA bf16 matrix pipe with f32 accumulation;
// aggregation uses one wave per edge with f32 global atomics.

typedef __bf16   v16bf __attribute__((ext_vector_type(16)));
typedef float    v8f   __attribute__((ext_vector_type(8)));
typedef unsigned v8u   __attribute__((ext_vector_type(8)));

#define IN_DIM 1433
#define HID    64
#define OUTC   7

// round-to-nearest-even f32x2 -> packed bf16x2 (pure integer ops, writes a
// whole dword: avoids half-register shuffles)
__device__ __forceinline__ unsigned pk2bf(float f0, float f1) {
  unsigned u0 = __builtin_bit_cast(unsigned, f0);
  unsigned u1 = __builtin_bit_cast(unsigned, f1);
  u0 = u0 + 0x7FFFu + ((u0 >> 16) & 1u);
  u1 = u1 + 0x7FFFu + ((u1 >> 16) & 1u);
  return (u0 >> 16) | (u1 & 0xFFFF0000u);
}

__device__ __forceinline__ unsigned short f2bf_u16(float f) {
  unsigned u = __builtin_bit_cast(unsigned, f);
  u = u + 0x7FFFu + ((u >> 16) & 1u);
  return (unsigned short)(u >> 16);
}

// ---------------------------------------------------------------- degrees ---
__global__ void __launch_bounds__(256) k_deg_init(float* deg, int n) {
  int i = blockIdx.x * blockDim.x + threadIdx.x;
  if (i < n) deg[i] = 1.0f;  // self-loop contributes 1
}

__global__ void __launch_bounds__(256) k_deg_edges(const int* __restrict__ dst,
                                                   float* deg, int E) {
  int e = blockIdx.x * blockDim.x + threadIdx.x;
  if (e < E) atomicAdd(&deg[dst[e]], 1.0f);
}

__global__ void __launch_bounds__(256) k_deg_to_dinv(float* deg, int n) {
  int i = blockIdx.x * blockDim.x + threadIdx.x;
  if (i < n) deg[i] = rsqrtf(deg[i]);  // deg >= 1 always (self loops)
}

// ------------------------------------------------- pack W into WMMA B tiles -
// Layout: [((kb*4 + ntile)*32 + lane)*16 + j] bf16, matching the 16-bit
// B-operand striping of V_WMMA_F32_16X16X32_BF16 (ISA 7.12.2): lane holds
// column n = nt*16 + (lane&15), K offsets {half*8+j} and {16+half*8+j}.
__global__ void __launch_bounds__(256) k_pack_w(const float* __restrict__ W,
                                                unsigned short* __restrict__ out,
                                                int K, int KB) {
  int tid = blockIdx.x * blockDim.x + threadIdx.x;
  int total = KB * 4 * 32 * 16;
  if (tid >= total) return;
  int j    = tid & 15;
  int lane = (tid >> 4) & 31;
  int nt   = (tid >> 9) & 3;
  int kb   = tid >> 11;
  int half = lane >> 4;
  int koff = (j < 8) ? (half * 8 + j) : (16 + half * 8 + (j - 8));
  int k    = kb * 32 + koff;
  int n    = nt * 16 + (lane & 15);
  float v  = (k < K) ? W[(size_t)k * HID + n] : 0.0f;
  out[tid] = f2bf_u16(v);
}

// ------------------------------------------------------------- WMMA GEMM ----
// C[M x 64] = bf16(A'[M x K]) @ Bpacked, f32 accumulate.
// EPI=1: A' = relu(A + abias) fused into the A-operand load (layer-2 path).
// TAIL=1: K not a multiple of 32 -> one branch-free trailing k-block.
// 8 waves/block; each wave owns a 16x64 tile (4 x 16x16 f32 accumulators).
template <int EPI, int TAIL>
__global__ void __launch_bounds__(256)
k_gemm_bf16(const float* __restrict__ A, int lda, int K, int KB,
            const unsigned short* __restrict__ Bp,
            const float* __restrict__ abias,
            float* __restrict__ out, int M) {
  const int lane = threadIdx.x & 31;
  const int wave = threadIdx.x >> 5;
  const int row0 = (blockIdx.x * 8 + wave) * 16;
  if (row0 >= M) return;                       // wave-uniform: EXEC stays full
  const int half = lane >> 4;
  int r = row0 + (lane & 15);
  if (r >= M) r = M - 1;                       // clamp loads; stores guarded
  const float* __restrict__ arow = A + (size_t)r * lda;

  v8f acc[4];
#pragma unroll
  for (int nt = 0; nt < 4; ++nt)
#pragma unroll
    for (int j = 0; j < 8; ++j) acc[nt][j] = 0.0f;

  const int KBfull = TAIL ? (K >> 5) : KB;

  for (int kb = 0; kb < KBfull; ++kb) {
    const int kbase = kb * 32 + half * 8;
    __builtin_prefetch((const void*)(arow + kb * 32 + 64), 0, 1);

    float f[16];
#pragma unroll
    for (int j = 0; j < 8; ++j) {
      f[j]     = arow[kbase + j];
      f[8 + j] = arow[kbase + 16 + j];
    }
    if (EPI) {
#pragma unroll
      for (int j = 0; j < 8; ++j) {
        f[j]     = fmaxf(f[j]     + abias[kbase + j],      0.0f);
        f[8 + j] = fmaxf(f[8 + j] + abias[kbase + 16 + j], 0.0f);
      }
    }
    v8u pk;
#pragma unroll
    for (int d = 0; d < 4; ++d) {
      pk[d]     = pk2bf(f[2 * d],     f[2 * d + 1]);
      pk[4 + d] = pk2bf(f[8 + 2 * d], f[9 + 2 * d]);
    }
    v16bf a = __builtin_bit_cast(v16bf, pk);

    const v16bf* __restrict__ bpt =
        (const v16bf*)(Bp + ((size_t)(kb * 4) * 32 + lane) * 16);
    v16bf b0 = bpt[0];     // one clause of 8 x global_load_b128 (L2-resident)
    v16bf b1 = bpt[32];
    v16bf b2 = bpt[64];
    v16bf b3 = bpt[96];
    acc[0] = __builtin_amdgcn_wmma_f32_16x16x32_bf16(false, a, false, b0,
                                                     (short)0, acc[0], false, false);
    acc[1] = __builtin_amdgcn_wmma_f32_16x16x32_bf16(false, a, false, b1,
                                                     (short)0, acc[1], false, false);
    acc[2] = __builtin_amdgcn_wmma_f32_16x16x32_bf16(false, a, false, b2,
                                                     (short)0, acc[2], false, false);
    acc[3] = __builtin_amdgcn_wmma_f32_16x16x32_bf16(false, a, false, b3,
                                                     (short)0, acc[3], false, false);
  }

  if (TAIL && (K & 31)) {
    // Branch-free trailing k-block: clamp the index so every load is
    // unconditional (stays in-bounds), then zero OOB columns with selects.
    const int kb = KBfull;
    const int kbase = kb * 32 + half * 8;
    float f[16];
#pragma unroll
    for (int j = 0; j < 8; ++j) {
      int ka = kbase + j, kc = kbase + 16 + j;
      int kal = ka < K ? ka : K - 1;
      int kcl = kc < K ? kc : K - 1;
      float va = arow[kal];
      float vc = arow[kcl];
      f[j]     = (ka < K) ? va : 0.0f;   // v_cndmask, no EXEC branches
      f[8 + j] = (kc < K) ? vc : 0.0f;
    }
    v8u pk;
#pragma unroll
    for (int d = 0; d < 4; ++d) {
      pk[d]     = pk2bf(f[2 * d],     f[2 * d + 1]);
      pk[4 + d] = pk2bf(f[8 + 2 * d], f[9 + 2 * d]);
    }
    v16bf a = __builtin_bit_cast(v16bf, pk);
    const v16bf* __restrict__ bpt =
        (const v16bf*)(Bp + ((size_t)(kb * 4) * 32 + lane) * 16);
    v16bf b0 = bpt[0];
    v16bf b1 = bpt[32];
    v16bf b2 = bpt[64];
    v16bf b3 = bpt[96];
    acc[0] = __builtin_amdgcn_wmma_f32_16x16x32_bf16(false, a, false, b0,
                                                     (short)0, acc[0], false, false);
    acc[1] = __builtin_amdgcn_wmma_f32_16x16x32_bf16(false, a, false, b1,
                                                     (short)0, acc[1], false, false);
    acc[2] = __builtin_amdgcn_wmma_f32_16x16x32_bf16(false, a, false, b2,
                                                     (short)0, acc[2], false, false);
    acc[3] = __builtin_amdgcn_wmma_f32_16x16x32_bf16(false, a, false, b3,
                                                     (short)0, acc[3], false, false);
  }

#pragma unroll
  for (int nt = 0; nt < 4; ++nt) {
    const int col = nt * 16 + (lane & 15);
#pragma unroll
    for (int j = 0; j < 8; ++j) {
      int m = row0 + half * 8 + j;
      if (m < M) out[(size_t)m * HID + col] = acc[nt][j];
    }
  }
}

// --------------------------------------------- aggregation (A+I normalized) -
// agg[i,:] = dinv[i]^2 * xw[i,:]   (self-loop term doubles as zero-init)
__global__ void __launch_bounds__(256)
k_selfloop_init(const float* __restrict__ xw, const float* __restrict__ dinv,
                float* __restrict__ agg, int n64) {
  int t = blockIdx.x * blockDim.x + threadIdx.x;
  if (t >= n64) return;
  float dv = dinv[t >> 6];
  agg[t] = xw[t] * dv * dv;
}

// One wave per edge: 32 lanes x float2 = 64 features; scatter via f32 atomics.
__global__ void __launch_bounds__(256)
k_scatter_edges(const int* __restrict__ src, const int* __restrict__ dst,
                const float* __restrict__ dinv, const float* __restrict__ xw,
                float* __restrict__ agg, int E) {
  int wid  = (blockIdx.x * blockDim.x + threadIdx.x) >> 5;
  int lane = threadIdx.x & 31;
  if (wid >= E) return;                        // wave-uniform
  int s = src[wid], d = dst[wid];
  float nrm = dinv[s] * dinv[d];
  const float2* __restrict__ xp = (const float2*)(xw + (size_t)s * HID);
  float2 v = xp[lane];
  float* op = agg + (size_t)d * HID + lane * 2;
  atomicAdd(op,     v.x * nrm);
  atomicAdd(op + 1, v.y * nrm);
}

// --------------------------------------- output head: relu + GEMV + softmax -
__global__ void __launch_bounds__(256)
k_final_softmax(const float* __restrict__ agg, const float* __restrict__ b2,
                const float* __restrict__ Wout, const float* __restrict__ bout,
                float* __restrict__ out, int N) {
  int i = blockIdx.x * blockDim.x + threadIdx.x;
  if (i >= N) return;
  float h[HID];
#pragma unroll
  for (int f = 0; f < HID; ++f)
    h[f] = fmaxf(agg[(size_t)i * HID + f] + b2[f], 0.0f);

  float lg[OUTC];
#pragma unroll
  for (int c = 0; c < OUTC; ++c) {
    float s = bout[c];
#pragma unroll
    for (int f = 0; f < HID; ++f) s += h[f] * Wout[f * OUTC + c];
    lg[c] = s;
  }
  float m = lg[0];
#pragma unroll
  for (int c = 1; c < OUTC; ++c) m = fmaxf(m, lg[c]);
  float sum = 0.0f;
#pragma unroll
  for (int c = 0; c < OUTC; ++c) { lg[c] = __expf(lg[c] - m); sum += lg[c]; }
  float inv = 1.0f / sum;
#pragma unroll
  for (int c = 0; c < OUTC; ++c) out[(size_t)i * OUTC + c] = lg[c] * inv;
}

// ----------------------------------------------------------------- launch ---
extern "C" void kernel_launch(void* const* d_in, const int* in_sizes, int n_in,
                              void* d_out, int out_size, void* d_ws, size_t ws_size,
                              hipStream_t stream) {
  (void)n_in; (void)out_size; (void)ws_size;
  const float* x    = (const float*)d_in[0];
  const int*   ei   = (const int*)d_in[1];
  const float* W1   = (const float*)d_in[2];
  const float* b1   = (const float*)d_in[3];
  const float* W2   = (const float*)d_in[4];
  const float* b2   = (const float*)d_in[5];
  const float* Wout = (const float*)d_in[6];
  const float* bout = (const float*)d_in[7];
  float* out = (float*)d_out;

  const int N = in_sizes[0] / IN_DIM;   // 100000 nodes
  const int E = in_sizes[1] / 2;        // 1600000 edges
  const int* src = ei;
  const int* dst = ei + E;

  const int KB1 = (IN_DIM + 31) / 32;   // 45 k-blocks (44 full + tail)
  const int KB2 = (HID + 31) / 32;      // 2 k-blocks

  // workspace carve-up (256B aligned): dinv | W1packed | W2packed | bufA | bufB
  char* ws = (char*)d_ws;
  size_t off = 0;
  auto carve = [&](size_t bytes) {
    void* p = ws + off;
    off += (bytes + 255) & ~(size_t)255;
    return p;
  };
  float*          dinv = (float*)carve((size_t)N * 4);
  unsigned short* w1p  = (unsigned short*)carve((size_t)KB1 * 2048 * 2);
  unsigned short* w2p  = (unsigned short*)carve((size_t)KB2 * 2048 * 2);
  float*          bufA = (float*)carve((size_t)N * HID * 4);
  float*          bufB = (float*)carve((size_t)N * HID * 4);

  const int T = 256;
  const int n64 = N * HID;
  const int gemm_blocks = ((N + 15) / 16 + 7) / 8;  // 8 waves x 16 rows each

  // normalization
  k_deg_init   <<<(N + T - 1) / T, T, 0, stream>>>(dinv, N);
  k_deg_edges  <<<(E + T - 1) / T, T, 0, stream>>>(dst, dinv, E);
  k_deg_to_dinv<<<(N + T - 1) / T, T, 0, stream>>>(dinv, N);

  // weight packing for WMMA B operands
  k_pack_w<<<(KB1 * 2048 + T - 1) / T, T, 0, stream>>>(W1, w1p, IN_DIM, KB1);
  k_pack_w<<<(KB2 * 2048 + T - 1) / T, T, 0, stream>>>(W2, w2p, HID, KB2);

  // layer 1: xw = x @ W1 ; agg = Dhat (A+I) Dhat xw
  k_gemm_bf16<0, 1><<<gemm_blocks, T, 0, stream>>>(x, IN_DIM, IN_DIM, KB1, w1p,
                                                   nullptr, bufA, N);
  k_selfloop_init<<<(n64 + T - 1) / T, T, 0, stream>>>(bufA, dinv, bufB, n64);
  k_scatter_edges<<<(E + 7) / 8, T, 0, stream>>>(src, dst, dinv, bufA, bufB, E);

  // layer 2: h1 = relu(agg + b1) fused into A-load ; xw2 = h1 @ W2 ; aggregate
  k_gemm_bf16<1, 0><<<gemm_blocks, T, 0, stream>>>(bufB, HID, HID, KB2, w2p,
                                                   b1, bufA, N);
  k_selfloop_init<<<(n64 + T - 1) / T, T, 0, stream>>>(bufA, dinv, bufB, n64);
  k_scatter_edges<<<(E + 7) / 8, T, 0, stream>>>(src, dst, dinv, bufA, bufB, E);

  // head: relu(agg + b2) @ Wout + bout, softmax
  k_final_softmax<<<(N + T - 1) / T, T, 0, stream>>>(bufB, b2, Wout, bout, out, N);
}
